// LSTM_40802189312138
// MI455X (gfx1250) — matrix operand order repaired and checked by reading, not compiled, not run on hardware
//
#include <hip/hip_runtime.h>
#include <hip/hip_bf16.h>
#include <math.h>

// ---------------------------------------------------------------------------
// MI455X (gfx1250) LSTM stack.
//   Phase 1 (parallel):  weights -> bf16 transposed; xin = tanh(X@Wi+bi);
//                        Z0x = xin@W0 + b0   (big WMMA GEMMs)
//   Phase 2 (sequential):persistent kernel, 64 WGs x 8 wave32 (split-K via
//                        LDS), TDM prefetch of Z0x tiles, global atomic
//                        barrier between layer0/layer1/steps.
//   Phase 3:             out = tanh(h1@Wo + bo) -> f32 d_out [64,128]
// ---------------------------------------------------------------------------

typedef __bf16 bf16;
typedef __attribute__((ext_vector_type(16))) __bf16 v16bf;
typedef __attribute__((ext_vector_type(8)))  __bf16 v8bf;
typedef __attribute__((ext_vector_type(8)))  float  v8f;

// ---- fast transcendental path: gfx1250 has V_TANH_F32 --------------------
#if __has_builtin(__builtin_amdgcn_tanhf)
__device__ __forceinline__ float fast_tanh(float x) { return __builtin_amdgcn_tanhf(x); }
#elif __has_builtin(__builtin_amdgcn_tanh_f32)
__device__ __forceinline__ float fast_tanh(float x) { return __builtin_amdgcn_tanh_f32(x); }
#else
__device__ __forceinline__ float fast_tanh(float x) { return tanhf(x); }
#endif
__device__ __forceinline__ float fast_sigmoid(float x) {
  return 0.5f * fast_tanh(0.5f * x) + 0.5f;   // 1 TRANS op, no divergence
}

__device__ __forceinline__ v8f wmma_bf16(v16bf a, v16bf b, v8f c) {
  return __builtin_amdgcn_wmma_f32_16x16x32_bf16(false, a, false, b,
                                                 (short)0, c, false, false);
}

// A fragment: 16x32 bf16, A row-major.  lanes 0-15: rows M=0..15, K{0..7,16..23};
// lanes 16-31: K{8..15,24..31}.  -> two aligned 16B loads per lane.
__device__ __forceinline__ v16bf load_a_frag(const bf16* __restrict__ A,
                                             int lda, int m0, int k0, int lane) {
  int row = m0 + (lane & 15);
  int kb  = k0 + ((lane & 16) ? 8 : 0);
  const bf16* p = A + (size_t)row * lda + kb;
  union { v16bf v; v8bf h[2]; } u;
  u.h[0] = *(const v8bf*)(p);
  u.h[1] = *(const v8bf*)(p + 16);
  return u.v;
}

// B fragment: 32x16 bf16 with B stored TRANSPOSED [N, K] row-major.
// lanes 0-15 (N=lane) hold K{0..7,16..23}, lanes 16-31 hold K{8..15,24..31} -> 2x16B loads.
__device__ __forceinline__ v16bf load_b_frag(const bf16* __restrict__ Bt,
                                             int ldb, int n0, int k0, int lane) {
  int row = n0 + (lane & 15);
  int kb  = k0 + ((lane & 16) ? 8 : 0);
  const bf16* p = Bt + (size_t)row * ldb + kb;
  union { v16bf v; v8bf h[2]; } u;
  u.h[0] = *(const v8bf*)(p);
  u.h[1] = *(const v8bf*)(p + 16);
  return u.v;
}

// ---- Tensor Data Mover (this toolchain: 6-arg builtin, clang-23 style) ---
#if __has_builtin(__builtin_amdgcn_tensor_load_to_lds) && \
    __has_builtin(__builtin_amdgcn_s_wait_tensorcnt)
#define USE_TDM 1
typedef __attribute__((ext_vector_type(4))) unsigned int u32x4;
typedef __attribute__((ext_vector_type(8))) int i32x8;
typedef __attribute__((ext_vector_type(4))) int i32x4;

// 2D tile load: tile_rows x tile_cols bf16 elements, row stride `stride_e`
// elements, packed contiguously into LDS at byte offset lds_off (static LDS
// assumed to start at offset 0 - single __shared__ object in the kernel).
__device__ __forceinline__ void tdm_load_2d(unsigned lds_off, const void* gsrc,
                                            unsigned tile_cols, unsigned tile_rows,
                                            unsigned tensor_w, unsigned stride_e) {
  unsigned long long ga = (unsigned long long)(size_t)gsrc;
  u32x4 g0 = { 1u,                                   // count=1, user descriptor
               lds_off,                              // D#.lds_addr
               (unsigned)(ga & 0xFFFFFFFFu),         // global_addr[31:0]
               (unsigned)((ga >> 32) & 0x01FFFFFFu) | 0x80000000u }; // [56:32]|type=2
  i32x8 g1 = { (int)(1u << 16),                      // wg_mask=0, data_size=2B
               (int)((tensor_w & 0xFFFFu) << 16),    // tensor_dim0[15:0]
               (int)((tensor_w >> 16) | ((tile_rows & 0xFFFFu) << 16)),
               (int)((tile_rows >> 16) | (tile_cols << 16)),  // tile_dim0
               (int)tile_rows,                       // tile_dim1 (tile_dim2=0)
               (int)stride_e,                        // tensor_dim0_stride[31:0]
               0, 0 };
  i32x4 z4 = { 0, 0, 0, 0 };
  i32x8 z8 = { 0, 0, 0, 0, 0, 0, 0, 0 };
  __builtin_amdgcn_tensor_load_to_lds(g0, g1, z4, z4, z8, 0);
}
#else
#define USE_TDM 0
#endif

// ---------------------------------------------------------------------------
// Generic C = act(A @ Bt^T + bias).  A:[M,K] bf16 row-major, Bt:[N,K] bf16.
// Block = 128 thr (4 waves on M); wave computes 32x64 (2x4 tiles).
// ---------------------------------------------------------------------------
template <bool OUT_BF16, bool ACT_TANH>
__global__ void __launch_bounds__(128)
gemm_bt_kernel(const bf16* __restrict__ A, const bf16* __restrict__ Bt,
               const float* __restrict__ bias, void* __restrict__ out,
               int M, int N, int K) {
  const int lane = threadIdx.x & 31;
  const int wave = threadIdx.x >> 5;
  const int m0 = blockIdx.x * 128 + wave * 32;
  const int n0 = blockIdx.y * 64;
  if (m0 >= M) return;  // wave-uniform

  v8f acc[2][4] = {};
  for (int k0 = 0; k0 < K; k0 += 32) {
    v16bf a0 = load_a_frag(A, K, m0,      k0, lane);
    v16bf a1 = load_a_frag(A, K, m0 + 16, k0, lane);
#pragma unroll
    for (int nt = 0; nt < 4; ++nt) {
      v16bf b = load_b_frag(Bt, K, n0 + nt * 16, k0, lane);
      acc[0][nt] = wmma_bf16(a0, b, acc[0][nt]);
      acc[1][nt] = wmma_bf16(a1, b, acc[1][nt]);
    }
  }
  const int nloc = lane & 15;
  const int mh   = (lane & 16) ? 8 : 0;
#pragma unroll
  for (int mt = 0; mt < 2; ++mt) {
#pragma unroll
    for (int nt = 0; nt < 4; ++nt) {
      const int col = n0 + nt * 16 + nloc;
      const float bv = bias[col];
#pragma unroll
      for (int r = 0; r < 8; ++r) {
        const int row = m0 + mt * 16 + mh + r;
        float v = acc[mt][nt][r] + bv;
        if (ACT_TANH) v = fast_tanh(v);
        if (OUT_BF16) ((bf16*)out)[(size_t)row * N + col] = (bf16)v;
        else          ((float*)out)[(size_t)row * N + col] = v;
      }
    }
  }
}

// ---------------------------------------------------------------------------
// Persistent 2-layer LSTM scan.
// Grid = 64 blocks (one 16-col tile of H, all 4 gates of those columns).
// 256 thr = 8 wave32: wave = (khalf, mtile); split-K halves the sequential
// load chain and gives 2 waves/SIMD; partials reduced through LDS.
// ---------------------------------------------------------------------------
__device__ __forceinline__ void grid_barrier(unsigned* bar, unsigned nblocks) {
  __syncthreads();
  __threadfence();  // release stores to agent scope
  if (threadIdx.x == 0) {
    unsigned* cnt = bar;
    unsigned* gen = bar + 1;
    unsigned g = __hip_atomic_load(gen, __ATOMIC_RELAXED, __HIP_MEMORY_SCOPE_AGENT);
    unsigned a = __hip_atomic_fetch_add(cnt, 1u, __ATOMIC_ACQ_REL, __HIP_MEMORY_SCOPE_AGENT);
    if (a == nblocks - 1) {
      __hip_atomic_store(cnt, 0u, __ATOMIC_RELAXED, __HIP_MEMORY_SCOPE_AGENT);
      __hip_atomic_fetch_add(gen, 1u, __ATOMIC_RELEASE, __HIP_MEMORY_SCOPE_AGENT);
    } else {
      while (__hip_atomic_load(gen, __ATOMIC_ACQUIRE, __HIP_MEMORY_SCOPE_AGENT) == g)
        __builtin_amdgcn_s_sleep(2);
    }
  }
  __syncthreads();
  __threadfence();  // acquire: drop stale L0 lines
}

__global__ void __launch_bounds__(256)
lstm_scan_kernel(const bf16* __restrict__ Z0x,  // [T*64, 4096] x-path + b0
                 const bf16* __restrict__ U0t,  // [4096,1024] transposed
                 const bf16* __restrict__ W1t,  // [4096,1024]
                 const bf16* __restrict__ U1t,  // [4096,1024]
                 const float* __restrict__ b1,  // [4096]
                 bf16* __restrict__ h0,         // [2][64*1024] double buffer
                 bf16* __restrict__ h1,         // [2][64*1024]
                 float* __restrict__ c0,        // [64*1024]
                 float* __restrict__ c1,        // [64*1024]
                 unsigned* __restrict__ bar, int T) {
  constexpr int H = 1024;
  constexpr int BH = 64 * H;
  constexpr unsigned PART_BYTES = 16 * 32 * sizeof(v8f);   // 16 KB
  // single shared object so static-LDS byte offsets are known (base 0)
  __shared__ __align__(16) unsigned char smem[PART_BYTES + 4 * 64 * 16 * sizeof(bf16)];
  v8f*  partbuf = (v8f*)smem;                 // [16 tiles][32 lanes]
  bf16* zbuf    = (bf16*)(smem + PART_BYTES); // [4 gates][64 rows][16 cols]

  const int lane  = threadIdx.x & 31;
  const int wave  = threadIdx.x >> 5;
  const int mtile = wave & 3;
  const int khalf = wave >> 2;
  const int j0 = blockIdx.x * 16;
  const int m0 = mtile * 16;
  const int kbase = khalf * 512;
  const int nloc = lane & 15;
  const int mh   = (lane & 16) ? 8 : 0;
  const int col  = j0 + nloc;

  for (int t = 0; t < T; ++t) {
    const int cur = t & 1, nxt = cur ^ 1;

    // ---------------- layer 0: z0 = Z0x[t] + h0@U0 -------------------------
    {
#if USE_TDM
      if (wave == 0) {  // DMA the 4 gate tiles; overlaps with the WMMA loop
        const bf16* zt = Z0x + (size_t)t * 64 * 4096;
#pragma unroll
        for (int g = 0; g < 4; ++g)
          tdm_load_2d(PART_BYTES + g * (64 * 16 * 2), zt + g * H + j0,
                      /*cols*/16, /*rows*/64, /*tensor_w*/4096, /*stride*/4096);
      }
#endif
      const bf16* hA = h0 + (size_t)cur * BH;
      v8f acc[4] = {};
      for (int k0 = kbase; k0 < kbase + 512; k0 += 32) {
        v16bf a = load_a_frag(hA, H, m0, k0, lane);
#pragma unroll
        for (int g = 0; g < 4; ++g)
          acc[g] = wmma_bf16(a, load_b_frag(U0t, H, g * H + j0, k0, lane), acc[g]);
      }
      if (khalf) {
#pragma unroll
        for (int g = 0; g < 4; ++g) partbuf[(mtile * 4 + g) * 32 + lane] = acc[g];
      }
#if USE_TDM
      if (wave == 0) __builtin_amdgcn_s_wait_tensorcnt(0);
#endif
      __syncthreads();
      if (!khalf) {
#pragma unroll
        for (int g = 0; g < 4; ++g) acc[g] = acc[g] + partbuf[(mtile * 4 + g) * 32 + lane];
#if !USE_TDM
        const bf16* zr = Z0x + (size_t)t * 64 * 4096;
#endif
#pragma unroll
        for (int r = 0; r < 8; ++r) {
          const int row = m0 + mh + r;
#if USE_TDM
          float zi = acc[0][r] + (float)zbuf[0 * 1024 + row * 16 + nloc];
          float zf = acc[1][r] + (float)zbuf[1 * 1024 + row * 16 + nloc];
          float zc = acc[2][r] + (float)zbuf[2 * 1024 + row * 16 + nloc];
          float zo = acc[3][r] + (float)zbuf[3 * 1024 + row * 16 + nloc];
#else
          const size_t zb = (size_t)row * 4096 + col;
          float zi = acc[0][r] + (float)zr[zb + 0 * H];
          float zf = acc[1][r] + (float)zr[zb + 1 * H];
          float zc = acc[2][r] + (float)zr[zb + 2 * H];
          float zo = acc[3][r] + (float)zr[zb + 3 * H];
#endif
          float cold = c0[row * H + col];
          float cn = fast_sigmoid(zf) * cold + fast_sigmoid(zi) * fast_tanh(zc);
          float hn = fast_sigmoid(zo) * fast_tanh(cn);
          c0[row * H + col] = cn;
          h0[(size_t)nxt * BH + row * H + col] = (bf16)hn;
        }
      }
    }
    grid_barrier(bar, gridDim.x);

    // ---------------- layer 1: z1 = h0_new@W1 + h1@U1 + b1 -----------------
    {
      const bf16* A1 = h0 + (size_t)nxt * BH;
      const bf16* A2 = h1 + (size_t)cur * BH;
      v8f acc[4] = {};
      for (int k0 = kbase; k0 < kbase + 512; k0 += 32) {
        v16bf a1f = load_a_frag(A1, H, m0, k0, lane);
        v16bf a2f = load_a_frag(A2, H, m0, k0, lane);
#pragma unroll
        for (int g = 0; g < 4; ++g) {
          acc[g] = wmma_bf16(a1f, load_b_frag(W1t, H, g * H + j0, k0, lane), acc[g]);
          acc[g] = wmma_bf16(a2f, load_b_frag(U1t, H, g * H + j0, k0, lane), acc[g]);
        }
      }
      if (khalf) {
#pragma unroll
        for (int g = 0; g < 4; ++g) partbuf[(mtile * 4 + g) * 32 + lane] = acc[g];
      }
      __syncthreads();
      if (!khalf) {
#pragma unroll
        for (int g = 0; g < 4; ++g) acc[g] = acc[g] + partbuf[(mtile * 4 + g) * 32 + lane];
        const float b_i = b1[0 * H + col], b_f = b1[1 * H + col];
        const float b_c = b1[2 * H + col], b_o = b1[3 * H + col];
#pragma unroll
        for (int r = 0; r < 8; ++r) {
          const int row = m0 + mh + r;
          float zi = acc[0][r] + b_i;
          float zf = acc[1][r] + b_f;
          float zc = acc[2][r] + b_c;
          float zo = acc[3][r] + b_o;
          float cold = c1[row * H + col];
          float cn = fast_sigmoid(zf) * cold + fast_sigmoid(zi) * fast_tanh(zc);
          float hn = fast_sigmoid(zo) * fast_tanh(cn);
          c1[row * H + col] = cn;
          h1[(size_t)nxt * BH + row * H + col] = (bf16)hn;
        }
      }
    }
    grid_barrier(bar, gridDim.x);
  }
}

// ---------------------------------------------------------------------------
// Setup kernels
// ---------------------------------------------------------------------------
__global__ void zero_kernel(unsigned* __restrict__ p, size_t n) {
  size_t i = (size_t)blockIdx.x * blockDim.x + threadIdx.x;
  if (i < n) p[i] = 0u;
}

// src [K,N] f32 row-major -> dst [N,K] bf16 row-major (transposed for B-frags)
__global__ void cvt_transpose_kernel(const float* __restrict__ src,
                                     bf16* __restrict__ dst, int K, int N) {
  size_t i = (size_t)blockIdx.x * blockDim.x + threadIdx.x;
  if (i >= (size_t)K * N) return;
  int n = (int)(i % N), k = (int)(i / N);
  dst[(size_t)n * K + k] = (bf16)src[i];
}

__global__ void cvt_f32_bf16_kernel(const float* __restrict__ src,
                                    bf16* __restrict__ dst, size_t n) {
  size_t i = (size_t)blockIdx.x * blockDim.x + threadIdx.x;
  if (i < n) dst[i] = (bf16)src[i];
}

// ---------------------------------------------------------------------------
extern "C" void kernel_launch(void* const* d_in, const int* in_sizes, int n_in,
                              void* d_out, int out_size, void* d_ws, size_t ws_size,
                              hipStream_t stream) {
  (void)in_sizes; (void)n_in; (void)out_size; (void)ws_size;
  const float* x   = (const float*)d_in[0];
  const float* Wi  = (const float*)d_in[1];
  const float* bi  = (const float*)d_in[2];
  const float* W0  = (const float*)d_in[3];
  const float* U0  = (const float*)d_in[4];
  const float* b0  = (const float*)d_in[5];
  const float* W1  = (const float*)d_in[6];
  const float* U1  = (const float*)d_in[7];
  const float* b1  = (const float*)d_in[8];
  const float* Wo  = (const float*)d_in[9];
  const float* bo  = (const float*)d_in[10];

  constexpr int B = 64, T = 512, F = 512, DIN = 1024, H = 1024, DOUT = 128;
  constexpr int BT = B * T;          // 32768 rows
  constexpr int G4 = 4 * H;          // 4096

  // --- workspace layout (~405 MB) ---
  char* ws = (char*)d_ws;
  size_t off = 0;
  auto take = [&](size_t bytes) {
    char* p = ws + off;
    off = (off + bytes + 255) & ~(size_t)255;
    return p;
  };
  unsigned* bar = (unsigned*)take(256);
  bf16*  h0  = (bf16*)take((size_t)2 * B * H * sizeof(bf16));
  bf16*  h1  = (bf16*)take((size_t)2 * B * H * sizeof(bf16));
  float* c0  = (float*)take((size_t)B * H * sizeof(float));
  float* c1  = (float*)take((size_t)B * H * sizeof(float));
  size_t zero_dwords = off / 4;      // bar + states start at offset 0
  bf16* WiT = (bf16*)take((size_t)DIN * F * sizeof(bf16));
  bf16* W0t = (bf16*)take((size_t)G4 * DIN * sizeof(bf16));
  bf16* U0t = (bf16*)take((size_t)G4 * H * sizeof(bf16));
  bf16* W1t = (bf16*)take((size_t)G4 * H * sizeof(bf16));
  bf16* U1t = (bf16*)take((size_t)G4 * H * sizeof(bf16));
  bf16* Wot = (bf16*)take((size_t)DOUT * H * sizeof(bf16));
  bf16* xbf = (bf16*)take((size_t)BT * F * sizeof(bf16));
  bf16* xin = (bf16*)take((size_t)BT * DIN * sizeof(bf16));
  bf16* Z0x = (bf16*)take((size_t)BT * G4 * sizeof(bf16));

  // 1) zero barrier + h/c state (initial carries are zero)
  zero_kernel<<<(unsigned)((zero_dwords + 255) / 256), 256, 0, stream>>>(bar, zero_dwords);

  // 2) weights -> bf16 transposed [N,K]
  auto cvtT = [&](const float* s, bf16* d, int K, int N) {
    size_t n = (size_t)K * N;
    cvt_transpose_kernel<<<(unsigned)((n + 255) / 256), 256, 0, stream>>>(s, d, K, N);
  };
  cvtT(Wi, WiT, F, DIN);
  cvtT(W0, W0t, DIN, G4);
  cvtT(U0, U0t, H, G4);
  cvtT(W1, W1t, H, G4);
  cvtT(U1, U1t, H, G4);
  cvtT(Wo, Wot, H, DOUT);

  // 3) x -> bf16
  {
    size_t n = (size_t)BT * F;
    cvt_f32_bf16_kernel<<<(unsigned)((n + 255) / 256), 256, 0, stream>>>(x, xbf, n);
  }

  // 4) xin = tanh(x @ Wi + bi)   [32768,512]x[512,1024] -> bf16
  gemm_bt_kernel<true, true><<<dim3(BT / 128, DIN / 64), 128, 0, stream>>>(
      xbf, WiT, bi, xin, BT, DIN, F);

  // 5) Z0x = xin @ W0 + b0       [32768,1024]x[1024,4096] -> bf16
  gemm_bt_kernel<true, false><<<dim3(BT / 128, G4 / 64), 128, 0, stream>>>(
      xin, W0t, b0, Z0x, BT, G4, DIN);

  // 6) persistent sequential scan: 64 WGs x 8 wave32 (split-K through LDS)
  lstm_scan_kernel<<<H / 16, 256, 0, stream>>>(Z0x, U0t, W1t, U1t, b1,
                                               h0, h1, c0, c1, bar, T);

  // 7) out = tanh(h1_final @ Wo + bo) -> f32 d_out [64,128]
  const bf16* h1f = h1 + (size_t)(T & 1) * B * H;  // T even -> buffer 0
  gemm_bt_kernel<false, true><<<dim3(1, DOUT / 64), 128, 0, stream>>>(
      h1f, Wot, bo, d_out, B, DOUT, H);
}
